// SimpleSequentialModel_49735721288310
// MI455X (gfx1250) — matrix operand order repaired and checked by reading, not compiled, not run on hardware
//
#include <hip/hip_runtime.h>
#include <stdint.h>

// Problem constants (from reference)
#define VOCAB 32000
#define DIMS  512
#define BATCH 2
#define SEQ   2048
#define MROWS (BATCH * SEQ)   // 4096

typedef __attribute__((ext_vector_type(16))) _Float16 v16h;
typedef __attribute__((ext_vector_type(8)))  _Float16 v8h;
typedef __attribute__((ext_vector_type(4)))  _Float16 v4h;
typedef __attribute__((ext_vector_type(8)))  float    v8f;
typedef __attribute__((ext_vector_type(2)))  float    v2f;

// ---------------- CDNA5 async global->LDS path (guarded) ----------------
#if __has_builtin(__builtin_amdgcn_global_load_async_to_lds_b128)
#define USE_ASYNC 1
#else
#define USE_ASYNC 0
#endif

#if USE_ASYNC
// Builtin signature (from clang diagnostic): arg0 = AS1 pointer to
// 16-byte int vector, arg1 = AS3 pointer, arg2/arg3 = imm offset / cpol.
typedef int b128_t __attribute__((vector_size(16)));
typedef __attribute__((address_space(1))) b128_t b128_glob;
typedef __attribute__((address_space(3))) b128_t b128_lds;
#endif

__device__ __forceinline__ void copy16_g2l(const void* g, void* l) {
#if USE_ASYNC
  // generic->AS1 via integer; generic LDS ptr low 32 bits == LDS offset -> AS3
  __builtin_amdgcn_global_load_async_to_lds_b128(
      (b128_glob*)(uintptr_t)g,
      (b128_lds*)(uint32_t)(uintptr_t)l,
      0, 0);
#else
  *(float4*)l = *(const float4*)g;
#endif
}

__device__ __forceinline__ void wait_async() {
#if USE_ASYNC
#if __has_builtin(__builtin_amdgcn_s_wait_asynccnt)
  __builtin_amdgcn_s_wait_asynccnt(0);
#else
  asm volatile("s_wait_asynccnt 0" ::: "memory");
#endif
#endif
}

// ---------------- K0: Wo f32 -> f16 ----------------
__global__ void __launch_bounds__(256) wo_to_half_kernel(
    const float* __restrict__ src, _Float16* __restrict__ dst, int n4) {
  int i = blockIdx.x * blockDim.x + threadIdx.x;
  if (i >= n4) return;
  float4 v = ((const float4*)src)[i];
  v4h h = { (_Float16)v.x, (_Float16)v.y, (_Float16)v.z, (_Float16)v.w };
  ((v4h*)dst)[i] = h;
}

// ---------------- K1: V = emb_table[idx] @ Wv^T  (f32 WMMA 16x16x4) ------
// M=4096, N=512, K=512. 128x128 block tile, K chunks of 64 staged in LDS.
#define LDAF 68   // 64 + 4 floats pad (272B row -> bank shift of 4 dwords)
__global__ void __launch_bounds__(256) v_gemm_kernel(
    const int* __restrict__ idx, const float* __restrict__ emb,
    const float* __restrict__ Wv, float* __restrict__ V) {
  extern __shared__ float smf[];           // A: 128*LDAF | B: 128*LDAF
  float* As = smf;
  float* Bs = smf + 128 * LDAF;
  const int tid = threadIdx.x;
  const int lane = tid & 31, w = tid >> 5;
  const int g = lane >> 4, l16 = lane & 15;
  const int wm = w & 3, wn = w >> 2;       // 4 waves in M, 2 in N
  const int mbase = blockIdx.y * 128, nbase = blockIdx.x * 128;

  v8f acc[2][4] = {};
  for (int kc = 0; kc < 8; ++kc) {         // K = 8 * 64
    __syncthreads();                       // previous readers done
#pragma unroll
    for (int i = 0; i < 8; ++i) {          // 2048 float4 chunks / 256 thr
      int c = tid + i * 256;
      int r = c >> 4, c4 = c & 15;         // 16 float4 per 64-float row
      int arow = idx[mbase + r];
      copy16_g2l(&emb[(size_t)arow * DIMS + kc * 64 + c4 * 4],
                 &As[r * LDAF + c4 * 4]);
      copy16_g2l(&Wv[(size_t)(nbase + r) * DIMS + kc * 64 + c4 * 4],
                 &Bs[r * LDAF + c4 * 4]);
    }
    wait_async();
    __syncthreads();
#pragma unroll
    for (int kk = 0; kk < 16; ++kk) {      // 16 k-steps of 4
      int k0 = kk * 4;
      v2f a[2], b[4];
#pragma unroll
      for (int mi = 0; mi < 2; ++mi)
        a[mi] = *(const v2f*)&As[(wm * 32 + mi * 16 + l16) * LDAF + k0 + 2 * g];
#pragma unroll
      for (int ni = 0; ni < 4; ++ni)
        b[ni] = *(const v2f*)&Bs[(wn * 64 + ni * 16 + l16) * LDAF + k0 + 2 * g];
#pragma unroll
      for (int mi = 0; mi < 2; ++mi)
#pragma unroll
        for (int ni = 0; ni < 4; ++ni)
          acc[mi][ni] = __builtin_amdgcn_wmma_f32_16x16x4_f32(
              false, a[mi], false, b[ni], (short)0, acc[mi][ni], false, false);
    }
  }
#pragma unroll
  for (int mi = 0; mi < 2; ++mi)
#pragma unroll
    for (int ni = 0; ni < 4; ++ni)
#pragma unroll
      for (int r = 0; r < 8; ++r) {
        int row = mbase + wm * 32 + mi * 16 + 8 * g + r;
        int col = nbase + wn * 64 + ni * 16 + l16;
        V[(size_t)row * DIMS + col] = acc[mi][ni][r];
      }
}

// ---------------- K2: causal prefix mean, write avg as f16 --------------
// One wave32 per (b, d) column. Lane owns a 64-element chunk; chunk sums are
// scanned across the wave with __shfl_up (warpSize == 32 on gfx1250).
__global__ void __launch_bounds__(256) cumavg_kernel(
    const float* __restrict__ V, _Float16* __restrict__ avgh) {
  int gid = blockIdx.x * blockDim.x + threadIdx.x;
  int wave = gid >> 5;               // 0..1023
  int lane = gid & 31;
  int b = wave >> 9;                 // / 512
  int d = wave & 511;
  const float* col = V + (size_t)b * SEQ * DIMS + d;
  _Float16* oc = avgh + (size_t)b * SEQ * DIMS + d;
  int t0 = lane * 64;
  float s = 0.f;
  for (int i = 0; i < 64; ++i) s += col[(size_t)(t0 + i) * DIMS];
  float pre = s;
#pragma unroll
  for (int off = 1; off < 32; off <<= 1) {
    float n = __shfl_up(pre, off, 32);
    if (lane >= off) pre += n;
  }
  float acc = pre - s;               // exclusive prefix of chunk sums
  for (int i = 0; i < 64; ++i) {
    acc += col[(size_t)(t0 + i) * DIMS];
    oc[(size_t)(t0 + i) * DIMS] = (_Float16)(acc / (float)(t0 + i + 1));
  }
}

// ---------------- K3: out = avg @ Wo^T + bo  (f16 WMMA 16x16x32) --------
// M=4096, N=32000, K=512. 128x128 block tile, double-buffered K chunks of
// 128 halves in LDS, async global->LDS staging, coalesced f32 epilogue.
#define LDH  136                     // 128 + 8 halves pad (272B row)
#define BUFH (2 * 128 * LDH)         // halves per buffer (A tile + B tile)
__global__ void __launch_bounds__(256) out_gemm_kernel(
    const _Float16* __restrict__ Ah, const _Float16* __restrict__ Bh,
    const float* __restrict__ bo, float* __restrict__ out) {
  extern __shared__ _Float16 smh[];  // 2 buffers * (A 128*LDH | B 128*LDH)
  const int tid = threadIdx.x;
  const int lane = tid & 31, w = tid >> 5;
  const int g = lane >> 4, l16 = lane & 15;
  const int wm = w & 3, wn = w >> 2;
  const int mbase = blockIdx.y * 128, nbase = blockIdx.x * 128;

  v8f acc[2][4] = {};
  // prologue: chunk 0 -> buffer 0
  {
    _Float16* As = smh;
    _Float16* Bs = smh + 128 * LDH;
#pragma unroll
    for (int i = 0; i < 8; ++i) {
      int c = tid + i * 256;
      int r = c >> 4, c8 = c & 15;   // 16 x b128 per 128-half row
      copy16_g2l(&Ah[(size_t)(mbase + r) * DIMS + c8 * 8], &As[r * LDH + c8 * 8]);
      copy16_g2l(&Bh[(size_t)(nbase + r) * DIMS + c8 * 8], &Bs[r * LDH + c8 * 8]);
    }
    wait_async();
    __syncthreads();
  }
  for (int kc = 0; kc < 4; ++kc) {
    if (kc + 1 < 4) {                // prefetch next chunk into other buffer
      _Float16* As = smh + ((kc + 1) & 1) * BUFH;
      _Float16* Bs = As + 128 * LDH;
      int ko = (kc + 1) * 128;
#pragma unroll
      for (int i = 0; i < 8; ++i) {
        int c = tid + i * 256;
        int r = c >> 4, c8 = c & 15;
        copy16_g2l(&Ah[(size_t)(mbase + r) * DIMS + ko + c8 * 8], &As[r * LDH + c8 * 8]);
        copy16_g2l(&Bh[(size_t)(nbase + r) * DIMS + ko + c8 * 8], &Bs[r * LDH + c8 * 8]);
      }
    }
    const _Float16* As = smh + (kc & 1) * BUFH;
    const _Float16* Bs = As + 128 * LDH;
#pragma unroll
    for (int kk = 0; kk < 4; ++kk) { // 4 k-steps of 32 per chunk
      v16h a[2], b[4];
#pragma unroll
      for (int mi = 0; mi < 2; ++mi) {
        // ISA 16-bit A 16x32 layout: halves[0..7]=K[8g..8g+7],
        // halves[8..15]=K[16+8g..16+8g+7], row m = lane%16
        const _Float16* p = As + (wm * 32 + mi * 16 + l16) * LDH + kk * 32;
        v8h lo = *(const v8h*)(p + 8 * g);
        v8h hi = *(const v8h*)(p + 16 + 8 * g);
        a[mi] = __builtin_shufflevector(lo, hi, 0, 1, 2, 3, 4, 5, 6, 7,
                                        8, 9, 10, 11, 12, 13, 14, 15);
      }
#pragma unroll
      for (int ni = 0; ni < 4; ++ni) {
        // B 32x16: lane holds column n; halves[0..7]=K[8g..8g+7], halves[8..15]=K[16+8g..16+8g+7]
        const _Float16* p = Bs + (wn * 64 + ni * 16 + l16) * LDH + kk * 32;
        v8h lo = *(const v8h*)(p + 8 * g);
        v8h hi = *(const v8h*)(p + 16 + 8 * g);
        b[ni] = __builtin_shufflevector(lo, hi, 0, 1, 2, 3, 4, 5, 6, 7,
                                        8, 9, 10, 11, 12, 13, 14, 15);
      }
#pragma unroll
      for (int mi = 0; mi < 2; ++mi)
#pragma unroll
        for (int ni = 0; ni < 4; ++ni)
          acc[mi][ni] = __builtin_amdgcn_wmma_f32_16x16x32_f16(
              false, a[mi], false, b[ni], (short)0, acc[mi][ni], false, false);
    }
    wait_async();
    __syncthreads();
  }
  // ---- epilogue: bias + LDS transpose -> coalesced float4 stores ----
  // Each wave stages its 32x64 f32 tile in a private LDS region (stride 68
  // dwords; conflict-free), then writes rows back as float4 (512B/instr).
  float* tw = (float*)smh + w * (32 * 68);   // 8704B per wave, 69632B total
#pragma unroll
  for (int ni = 0; ni < 4; ++ni) {
    float bn = bo[nbase + wn * 64 + ni * 16 + l16];
#pragma unroll
    for (int mi = 0; mi < 2; ++mi)
#pragma unroll
      for (int r = 0; r < 8; ++r)
        tw[(mi * 16 + 8 * g + r) * 68 + ni * 16 + l16] = acc[mi][ni][r] + bn;
  }
  // LDS ops are in-order within a wave; each wave reads only its own region.
  const int rowbase = mbase + wm * 32;
  const int colbase = nbase + wn * 64;
#pragma unroll
  for (int i = 0; i < 16; ++i) {
    int id = i * 32 + lane;          // 0..511 float4 chunks of the 32x64 tile
    int r = id >> 4;                 // row 0..31
    int c4 = id & 15;                // float4 index within row
    float4 vv = *(const float4*)&tw[r * 68 + c4 * 4];
    *(float4*)&out[(size_t)(rowbase + r) * VOCAB + colbase + c4 * 4] = vv;
  }
}

// ---------------- launch ----------------
extern "C" void kernel_launch(void* const* d_in, const int* in_sizes, int n_in,
                              void* d_out, int out_size, void* d_ws, size_t ws_size,
                              hipStream_t stream) {
  (void)in_sizes; (void)n_in; (void)out_size; (void)ws_size;
  const int*   idx = (const int*)d_in[0];
  const float* emb = (const float*)d_in[1];
  const float* Wv  = (const float*)d_in[2];
  const float* Wo  = (const float*)d_in[3];
  const float* bo  = (const float*)d_in[4];
  float* out = (float*)d_out;

  // workspace layout (~45.4 MB): V f32 | avg f16 | Wo f16
  char* ws = (char*)d_ws;
  float*    V    = (float*)ws;                                   //  8,388,608 B
  _Float16* avgh = (_Float16*)(ws + 8388608);                    //  4,194,304 B
  _Float16* Woh  = (_Float16*)(ws + 8388608 + 4194304);          // 32,768,000 B

  int n4 = VOCAB * DIMS / 4;
  wo_to_half_kernel<<<(n4 + 255) / 256, 256, 0, stream>>>(Wo, Woh, n4);
  v_gemm_kernel<<<dim3(4, 32), 256, 2 * 128 * LDAF * sizeof(float), stream>>>(
      idx, emb, Wv, V);
  cumavg_kernel<<<128, 256, 0, stream>>>(V, avgh);
  out_gemm_kernel<<<dim3(250, 32), 256, 2 * BUFH * sizeof(_Float16), stream>>>(
      avgh, Woh, bo, out);
}
